// STULayer_3332894622428
// MI455X (gfx1250) — compile-verified, matrix operand order checked by reading, not run on hardware
//
#include <hip/hip_runtime.h>
#include <hip/hip_bf16.h>

// ---------------------------------------------------------------------------
// STU layer for MI455X (gfx1250): bf16 WMMA pipeline with f32 accumulation.
// Tensor Data Mover stages the GEMM operand tiles and attention Q/K tiles into LDS; the
// GEMM K-loop double-buffers TDM against WMMA (TENSORcnt ping-pong).
// ---------------------------------------------------------------------------

typedef __bf16 bf16;
typedef __attribute__((ext_vector_type(8)))  __bf16 bf16x8;
typedef __attribute__((ext_vector_type(16))) __bf16 bf16x16;
typedef __attribute__((ext_vector_type(8)))  float  f32x8;
typedef __attribute__((ext_vector_type(4)))  unsigned int u32x4;
typedef __attribute__((ext_vector_type(8)))  int i32x8;
typedef __attribute__((ext_vector_type(4)))  int i32x4;

#define DEVI __device__ __forceinline__

constexpr int   B_   = 4;
constexpr int   S_   = 2048;
constexpr int   E_   = 1024;
constexpr int   H_   = 8;
constexpr int   HD_  = 128;
constexpr int   AD_  = 128;
constexpr int   T_   = B_ * S_;                  // 8192 tokens
constexpr int   N1_  = (HD_ * 2 + AD_ * 2) * H_; // 4096 (uvqk width)
constexpr int   K2_  = 3 * H_ * HD_;             // 3072 (proj width)
constexpr float EPSV  = 1e-6f;
constexpr float ALPHA = 0.08838834764831845f;    // 1/sqrt(128)
constexpr float INV_S = 1.0f / 2048.0f;

DEVI float sigmoidf_(float x) { return 1.0f / (1.0f + __expf(-x)); }

DEVI f32x8 wmma_bf16(bf16x16 a, bf16x16 b, f32x8 c) {
  return __builtin_amdgcn_wmma_f32_16x16x32_bf16(
      /*neg_a=*/false, a, /*neg_b=*/false, b,
      /*c_mod=*/(short)0, c, /*reuse_a=*/false, /*reuse_b=*/false);
}

// Load a 16x32 bf16 operand fragment from LDS stored [row][Kdim] (row-major,
// `stride` bf16 elements per row). Per ISA 16-bit A layout:
//   lanes 0-15 : M = lane,    elems 0-7 -> K 0..7,  elems 8-15 -> K 16..23
//   lanes 16-31: M = lane-16, K offset +8
// Both halves are contiguous 16-byte runs -> two ds_load_b128 per fragment.
DEVI bf16x16 load_frag(const bf16* base, int row, int stride, int koff) {
  const int lane = threadIdx.x & 31;
  const bf16* p = base + (row + (lane & 15)) * stride + koff + ((lane >> 4) << 3);
  bf16x8 lo = *(const bf16x8*)(p);
  bf16x8 hi = *(const bf16x8*)(p + 16);
  bf16x16 f;
#pragma unroll
  for (int i = 0; i < 8; ++i) { f[i] = lo[i]; f[i + 8] = hi[i]; }
  return f;
}

// ---------------------------------------------------------------------------
// Tensor Data Mover: DMA a [tile_d1 rows x tile_d0 elems] bf16 tile from a
// row-major global tensor (row stride `row_elems`) into LDS, stored
// contiguously as [tile_d1][tile_d0].  D# per CDNA5 ISA ch.8.
// ---------------------------------------------------------------------------
DEVI void tdm_load_2d(const bf16* tile_gaddr, unsigned lds_off,
                      unsigned tile_d0, unsigned tile_d1, unsigned row_elems) {
  const unsigned long long ga = (unsigned long long)(size_t)tile_gaddr;
  u32x4 g0;
  g0[0] = 1u;                                            // count=1, user D#
  g0[1] = lds_off;                                       // lds_addr (bytes)
  g0[2] = (unsigned)(ga & 0xFFFFFFFFu);                  // global_addr[31:0]
  g0[3] = (unsigned)((ga >> 32) & 0x1FFFFFFu)            // global_addr[56:32]
          | (2u << 30);                                  // type=2 ("image")
  i32x8 g1;
  g1[0] = (int)(1u << 16);                               // data_size=1 (2B)
  g1[1] = (int)((row_elems & 0xFFFFu) << 16);            // tensor_dim0[15:0]
  g1[2] = (int)((row_elems >> 16) & 0xFFFFu);            // dim0[31:16], dim1[15:0]=0
  g1[3] = (int)(0x10u | (tile_d0 << 16));                // tensor_dim1=2^20, tile_dim0
  g1[4] = (int)tile_d1;                                  // tile_dim1, tile_dim2=0
  g1[5] = (int)row_elems;                                // tensor_dim0_stride[31:0]
  g1[6] = 0;                                             // stride[47:32], dim1_stride
  g1[7] = 0;
  i32x4 z4 = {0, 0, 0, 0};
#if __clang_major__ >= 23
  i32x8 z8 = {0, 0, 0, 0, 0, 0, 0, 0};
  __builtin_amdgcn_tensor_load_to_lds(g0, g1, z4, z4, z8, 0);
#else
  __builtin_amdgcn_tensor_load_to_lds(g0, g1, z4, z4, 0);
#endif
}

// ---------------------------------------------------------------------------
// Block/warp reduction helpers (wave32).
// ---------------------------------------------------------------------------
DEVI void block_reduce2(float& a, float& b, float* s1, float* s2) {
  const int lane = threadIdx.x & 31;
  const int w    = threadIdx.x >> 5;
#pragma unroll
  for (int o = 16; o > 0; o >>= 1) {
    a += __shfl_xor(a, o, 32);
    b += __shfl_xor(b, o, 32);
  }
  if (lane == 0) { s1[w] = a; s2[w] = b; }
  __syncthreads();
  if (threadIdx.x == 0) {
    float ra = 0.f, rb = 0.f;
#pragma unroll
    for (int i = 0; i < 8; ++i) { ra += s1[i]; rb += s2[i]; }
    s1[0] = ra; s2[0] = rb;
  }
  __syncthreads();
  a = s1[0]; b = s2[0];
}

// ---------------------------------------------------------------------------
// Kernel 1: input LayerNorm, fp32 -> bf16.  One block per token row (E=1024).
// ---------------------------------------------------------------------------
__global__ __launch_bounds__(256) void k_ln_in(const float* __restrict__ x,
                                               const float* __restrict__ scale,
                                               const float* __restrict__ bias,
                                               bf16* __restrict__ xn) {
  __shared__ float s1[8], s2[8];
  const int row = blockIdx.x;
  const float* xr = x + (size_t)row * E_;
  float vals[4], sum = 0.f, sq = 0.f;
#pragma unroll
  for (int i = 0; i < 4; ++i) {
    float v = xr[threadIdx.x + i * 256];
    vals[i] = v; sum += v; sq += v * v;
  }
  block_reduce2(sum, sq, s1, s2);
  const float mu   = sum * (1.0f / E_);
  const float var  = sq * (1.0f / E_) - mu * mu;
  const float rstd = rsqrtf(var + EPSV);
#pragma unroll
  for (int i = 0; i < 4; ++i) {
    int c = threadIdx.x + i * 256;
    xn[(size_t)row * E_ + c] = (bf16)((vals[i] - mu) * rstd * scale[c] + bias[c]);
  }
}

// ---------------------------------------------------------------------------
// Kernel 2: one-shot weight convert+transpose: f32 [R][C] -> bf16 [C][R].
// ---------------------------------------------------------------------------
__global__ __launch_bounds__(256) void k_transpose(const float* __restrict__ in,
                                                   bf16* __restrict__ out,
                                                   int R, int C) {
  int idx = blockIdx.x * 256 + threadIdx.x;
  if (idx < R * C) {
    int r = idx / C, c = idx - r * C;
    out[(size_t)c * R + r] = (bf16)in[idx];
  }
}

// ---------------------------------------------------------------------------
// Shared GEMM core: C[128x128] += A[m0..][K] * Bt[n0..][K]^T, bf16 in, f32 acc.
// 256 threads = 8 waves arranged 4(M)x2(N); each wave owns 32x64 = 2x4 tiles.
// Double-buffered TDM: wave 0 issues the next tile-pair into the alternate
// LDS buffer, then waits TENSORcnt<=2 so the current pair (in-order) is done
// while the next pair streams in behind the WMMA work.
// As/Bs are 2 x [128][32] bf16 buffers (ping-pong).
// ---------------------------------------------------------------------------
DEVI void gemm_core(const bf16* __restrict__ A, const bf16* __restrict__ Bt,
                    int K, int m0, int n0, bf16* As, bf16* Bs,
                    f32x8 acc[2][4]) {
  const int wave = threadIdx.x >> 5;
  const int wm   = (wave >> 1) * 32;
  const int wn   = (wave & 1) * 64;
  const unsigned lds_a = (unsigned)(size_t)(void*)As;
  const unsigned lds_b = (unsigned)(size_t)(void*)Bs;
#pragma unroll
  for (int i = 0; i < 2; ++i)
#pragma unroll
    for (int j = 0; j < 4; ++j)
      acc[i][j] = (f32x8){0.f, 0.f, 0.f, 0.f, 0.f, 0.f, 0.f, 0.f};

  const int nsteps = K / 32;
  if (wave == 0) {
    tdm_load_2d(A  + (size_t)m0 * K, lds_a, 32, 128, (unsigned)K);
    tdm_load_2d(Bt + (size_t)n0 * K, lds_b, 32, 128, (unsigned)K);
  }
  for (int i = 0; i < nsteps; ++i) {
    const int cur = i & 1;
    if (wave == 0) {
      if (i + 1 < nsteps) {
        const unsigned nb = (unsigned)((i + 1) & 1) * 8192u;  // bytes
        const int k1 = (i + 1) * 32;
        tdm_load_2d(A  + (size_t)m0 * K + k1, lds_a + nb, 32, 128, (unsigned)K);
        tdm_load_2d(Bt + (size_t)n0 * K + k1, lds_b + nb, 32, 128, (unsigned)K);
        __builtin_amdgcn_s_wait_tensorcnt(2);  // current pair has landed
      } else {
        __builtin_amdgcn_s_wait_tensorcnt(0);
      }
    }
    __syncthreads();  // current tiles visible to all waves
    const bf16* Ac = As + cur * 4096;
    const bf16* Bc = Bs + cur * 4096;
    bf16x16 af[2], bfr[4];
#pragma unroll
    for (int ii = 0; ii < 2; ++ii) af[ii] = load_frag(Ac, wm + ii * 16, 32, 0);
#pragma unroll
    for (int j = 0; j < 4; ++j) bfr[j] = load_frag(Bc, wn + j * 16, 32, 0);
#pragma unroll
    for (int ii = 0; ii < 2; ++ii)
#pragma unroll
      for (int j = 0; j < 4; ++j)
        acc[ii][j] = wmma_bf16(af[ii], bfr[j], acc[ii][j]);
    __syncthreads();  // reads done: buffer reusable by the i+2 DMA
  }
}

// ---------------------------------------------------------------------------
// Kernel 3: uvqk GEMM (8192x1024)x(1024x4096) + beta, split & activate.
// The u/v/q/k section is uniform per 16-wide N tile -> branch at tile scope.
// ---------------------------------------------------------------------------
__global__ __launch_bounds__(256) void k_gemm_uvqk(
    const bf16* __restrict__ xn, const bf16* __restrict__ w1t,
    const float* __restrict__ beta, float* __restrict__ u,
    bf16* __restrict__ v, bf16* __restrict__ q, bf16* __restrict__ k) {
  __shared__ __align__(16) bf16 As[2 * 128 * 32];
  __shared__ __align__(16) bf16 Bs[2 * 128 * 32];
  const int m0 = blockIdx.x * 128;
  const int n0 = blockIdx.y * 128;
  f32x8 acc[2][4];
  gemm_core(xn, w1t, E_, m0, n0, As, Bs, acc);

  const int lane = threadIdx.x & 31;
  const int wave = threadIdx.x >> 5;
  const int wm = (wave >> 1) * 32, wn = (wave & 1) * 64;
  const int hh = (lane >> 4) << 3, ln = lane & 15;
  const int tbase = m0 + wm + hh;
#pragma unroll
  for (int tj = 0; tj < 4; ++tj) {
    const int   nb  = n0 + wn + tj * 16;     // uniform across the wave tile
    const int   sec = nb >> 10;              // 0:u 1:v 2:q 3:k (uniform)
    const float bt  = beta[nb + ln];
    const size_t cb = (size_t)((nb & 1023) + ln);
    if (sec == 0) {
#pragma unroll
      for (int ti = 0; ti < 2; ++ti)
#pragma unroll
        for (int i = 0; i < 8; ++i) {
          float val = acc[ti][tj][i] + bt;
          u[(size_t)(tbase + ti * 16 + i) * 1024 + cb] = val * sigmoidf_(val);
        }
    } else if (sec == 1) {
#pragma unroll
      for (int ti = 0; ti < 2; ++ti)
#pragma unroll
        for (int i = 0; i < 8; ++i)
          v[(size_t)(tbase + ti * 16 + i) * 1024 + cb] = (bf16)(acc[ti][tj][i] + bt);
    } else if (sec == 2) {
#pragma unroll
      for (int ti = 0; ti < 2; ++ti)
#pragma unroll
        for (int i = 0; i < 8; ++i)
          q[(size_t)(tbase + ti * 16 + i) * 1024 + cb] = (bf16)(acc[ti][tj][i] + bt);
    } else {
#pragma unroll
      for (int ti = 0; ti < 2; ++ti)
#pragma unroll
        for (int i = 0; i < 8; ++i)
          k[(size_t)(tbase + ti * 16 + i) * 1024 + cb] = (bf16)(acc[ti][tj][i] + bt);
    }
  }
}

// ---------------------------------------------------------------------------
// Kernel 4: attention.  One block per (b, h, 64-query tile); loops over 32
// 64-wide key blocks.  Q and K tiles are TDM-DMA'd into LDS (the K DMA
// overlaps the all-wave V load+transpose); scores = silu(Q K^T * ALPHA)/S
// with the target-aware mask, then W @ V accumulated in f32.
// ---------------------------------------------------------------------------
__global__ __launch_bounds__(256) void k_attn(
    const bf16* __restrict__ q, const bf16* __restrict__ kk,
    const bf16* __restrict__ v, const int* __restrict__ num_targets,
    float* __restrict__ attn) {
  __shared__ __align__(16) bf16 Qs[64 * 128];
  __shared__ __align__(16) bf16 Ks[64 * 128];
  __shared__ __align__(16) bf16 Vt[128 * 64];   // V^T: [d][krow]
  __shared__ __align__(16) bf16 Ws[64 * 64];    // masked silu weights (bf16)

  const int b  = blockIdx.z, h = blockIdx.y;
  const int q0 = blockIdx.x * 64;
  const int tid = threadIdx.x, wave = tid >> 5, lane = tid & 31;
  const int maxid = S_ - num_targets[b];
  const int qr = (wave >> 1) * 16;   // wave's query-row group (stage 1 & 2)
  const int kg = (wave & 1) * 32;    // stage-1 key-column group
  const int dg = (wave & 1) * 64;    // stage-2 output-d group
  const int hh = (lane >> 4) << 3, ln = lane & 15;
  const unsigned lds_q = (unsigned)(size_t)(void*)Qs;
  const unsigned lds_k = (unsigned)(size_t)(void*)Ks;

  // DMA the Q tile (64 rows x 128 bf16, row stride 1024) once.
  if (wave == 0)
    tdm_load_2d(q + (size_t)(b * S_ + q0) * 1024 + h * 128, lds_q, 128, 64, 1024);

  f32x8 acc2[4];
#pragma unroll
  for (int td = 0; td < 4; ++td)
    acc2[td] = (f32x8){0.f, 0.f, 0.f, 0.f, 0.f, 0.f, 0.f, 0.f};

  for (int kb = 0; kb < S_ / 64; ++kb) {
    const int k0 = kb * 64;
    __syncthreads();  // prior iteration finished reading Ks/Vt/Ws
    if (wave == 0)
      tdm_load_2d(kk + (size_t)(b * S_ + k0) * 1024 + h * 128, lds_k, 128, 64, 1024);
    // All waves: load V tile and transpose into Vt while the K DMA streams.
#pragma unroll
    for (int it = 0; it < 4; ++it) {
      int chunk = tid + it * 256;
      int r = chunk >> 4, c8 = (chunk & 15) * 8;
      bf16x8 vv = *(const bf16x8*)(v + (size_t)(b * S_ + k0 + r) * 1024 + h * 128 + c8);
#pragma unroll
      for (int j = 0; j < 8; ++j) Vt[(c8 + j) * 64 + r] = vv[j];
    }
    if (wave == 0) __builtin_amdgcn_s_wait_tensorcnt(0);  // Q (+K) landed
    __syncthreads();

    // Stage 1: scores tile (wave: 16 q-rows x 32 k-cols).
    f32x8 sc[2];
#pragma unroll
    for (int tj = 0; tj < 2; ++tj)
      sc[tj] = (f32x8){0.f, 0.f, 0.f, 0.f, 0.f, 0.f, 0.f, 0.f};
#pragma unroll
    for (int dk = 0; dk < 4; ++dk) {
      bf16x16 aq = load_frag(Qs, qr, 128, dk * 32);
#pragma unroll
      for (int tj = 0; tj < 2; ++tj) {
        bf16x16 bk = load_frag(Ks, kg + tj * 16, 128, dk * 32);
        sc[tj] = wmma_bf16(aq, bk, sc[tj]);
      }
    }
    // silu + scale + target-aware causal mask -> Ws (bf16)
#pragma unroll
    for (int tj = 0; tj < 2; ++tj)
#pragma unroll
      for (int i = 0; i < 8; ++i) {
        int qi = q0 + qr + i + hh;
        int ki = k0 + kg + tj * 16 + ln;
        float s = sc[tj][i] * ALPHA;
        float w = s * sigmoidf_(s) * INV_S;
        int iq = min(qi, maxid), ik = min(ki, maxid);
        if (!((qi == ki) || (iq > ik))) w = 0.f;
        Ws[(qr + i + hh) * 64 + kg + tj * 16 + ln] = (bf16)w;
      }
    __syncthreads();

    // Stage 2: attn(16 q-rows x 64 d) += W(16x64) @ V(64x64-slice).
#pragma unroll
    for (int dk = 0; dk < 2; ++dk) {
      bf16x16 aw = load_frag(Ws, qr, 64, dk * 32);
#pragma unroll
      for (int td = 0; td < 4; ++td) {
        bf16x16 bv = load_frag(Vt, dg + td * 16, 64, dk * 32);
        acc2[td] = wmma_bf16(aw, bv, acc2[td]);
      }
    }
  }

#pragma unroll
  for (int td = 0; td < 4; ++td)
#pragma unroll
    for (int i = 0; i < 8; ++i) {
      int t = b * S_ + q0 + qr + i + hh;
      int d = h * 128 + dg + td * 16 + ln;
      attn[(size_t)t * 1024 + d] = acc2[td][i];
    }
}

// ---------------------------------------------------------------------------
// Kernel 5: output LayerNorm + gating; build bf16 proj = [u, attn, u*ln(attn)].
// ---------------------------------------------------------------------------
__global__ __launch_bounds__(256) void k_ln_out(const float* __restrict__ attn,
                                                const float* __restrict__ u,
                                                const float* __restrict__ scale,
                                                const float* __restrict__ bias,
                                                bf16* __restrict__ proj) {
  __shared__ float s1[8], s2[8];
  const int row = blockIdx.x;
  const float* ar = attn + (size_t)row * 1024;
  const float* ur = u + (size_t)row * 1024;
  float vals[4], sum = 0.f, sq = 0.f;
#pragma unroll
  for (int i = 0; i < 4; ++i) {
    float a = ar[threadIdx.x + i * 256];
    vals[i] = a; sum += a; sq += a * a;
  }
  block_reduce2(sum, sq, s1, s2);
  const float mu   = sum * (1.0f / 1024.0f);
  const float var  = sq * (1.0f / 1024.0f) - mu * mu;
  const float rstd = rsqrtf(var + EPSV);
  bf16* pr = proj + (size_t)row * K2_;
#pragma unroll
  for (int i = 0; i < 4; ++i) {
    int c = threadIdx.x + i * 256;
    float a  = vals[i];
    float nu = ur[c];
    float na = (a - mu) * rstd * scale[c] + bias[c];
    pr[c]        = (bf16)nu;          // u
    pr[1024 + c] = (bf16)a;           // attn
    pr[2048 + c] = (bf16)(nu * na);   // gated
  }
}

// ---------------------------------------------------------------------------
// Kernel 6: output GEMM (8192x3072)x(3072x1024) + residual x -> d_out (f32).
// ---------------------------------------------------------------------------
__global__ __launch_bounds__(256) void k_gemm_out(
    const bf16* __restrict__ proj, const bf16* __restrict__ w2t,
    const float* __restrict__ x, float* __restrict__ out) {
  __shared__ __align__(16) bf16 As[2 * 128 * 32];
  __shared__ __align__(16) bf16 Bs[2 * 128 * 32];
  const int m0 = blockIdx.x * 128;
  const int n0 = blockIdx.y * 128;
  f32x8 acc[2][4];
  gemm_core(proj, w2t, K2_, m0, n0, As, Bs, acc);

  const int lane = threadIdx.x & 31;
  const int wave = threadIdx.x >> 5;
  const int wm = (wave >> 1) * 32, wn = (wave & 1) * 64;
  const int hh = (lane >> 4) << 3, ln = lane & 15;
#pragma unroll
  for (int ti = 0; ti < 2; ++ti)
#pragma unroll
    for (int tj = 0; tj < 4; ++tj)
#pragma unroll
      for (int i = 0; i < 8; ++i) {
        int t = m0 + wm + ti * 16 + i + hh;
        int n = n0 + wn + tj * 16 + ln;
        size_t off = (size_t)t * E_ + n;
        out[off] = x[off] + acc[ti][tj][i];
      }
}

// ---------------------------------------------------------------------------
// Host-side launch.
// ---------------------------------------------------------------------------
extern "C" void kernel_launch(void* const* d_in, const int* in_sizes, int n_in,
                              void* d_out, int out_size, void* d_ws, size_t ws_size,
                              hipStream_t stream) {
  const float* x          = (const float*)d_in[0];
  const int*   num_tgt    = (const int*)d_in[1];
  const float* uvqk_w     = (const float*)d_in[2];
  const float* uvqk_beta  = (const float*)d_in[3];
  const float* out_w      = (const float*)d_in[4];
  const float* in_scale   = (const float*)d_in[5];
  const float* in_bias    = (const float*)d_in[6];
  const float* out_scale  = (const float*)d_in[7];
  const float* out_bias   = (const float*)d_in[8];
  float*       out        = (float*)d_out;

  char* wsp = (char*)d_ws;
  auto take = [&](size_t bytes) -> void* {
    void* p = (void*)wsp;
    wsp += (bytes + 255) & ~(size_t)255;
    return p;
  };
  bf16*  xn   = (bf16*) take((size_t)T_ * E_ * 2);
  bf16*  w1t  = (bf16*) take((size_t)N1_ * E_ * 2);
  bf16*  w2t  = (bf16*) take((size_t)E_ * K2_ * 2);
  float* u    = (float*)take((size_t)T_ * 1024 * 4);
  bf16*  vb   = (bf16*) take((size_t)T_ * 1024 * 2);
  bf16*  qb   = (bf16*) take((size_t)T_ * 1024 * 2);
  bf16*  kb   = (bf16*) take((size_t)T_ * 1024 * 2);
  float* attn = (float*)take((size_t)T_ * 1024 * 4);
  bf16*  proj = (bf16*) take((size_t)T_ * K2_ * 2);

  k_ln_in<<<T_, 256, 0, stream>>>(x, in_scale, in_bias, xn);
  k_transpose<<<(E_ * N1_ + 255) / 256, 256, 0, stream>>>(uvqk_w, w1t, E_, N1_);
  k_transpose<<<(K2_ * E_ + 255) / 256, 256, 0, stream>>>(out_w, w2t, K2_, E_);
  k_gemm_uvqk<<<dim3(T_ / 128, N1_ / 128), 256, 0, stream>>>(
      xn, w1t, uvqk_beta, u, vb, qb, kb);
  k_attn<<<dim3(S_ / 64, H_, B_), 256, 0, stream>>>(qb, kb, vb, num_tgt, attn);
  k_ln_out<<<T_, 256, 0, stream>>>(attn, u, out_scale, out_bias, proj);
  k_gemm_out<<<dim3(T_ / 128, E_ / 128), 256, 0, stream>>>(proj, w2t, x, out);
}
